// NTXentLoss_44452911514244
// MI455X (gfx1250) — matrix operand order repaired and checked by reading, not compiled, not run on hardware
//
#include <hip/hip_runtime.h>

#define BATCH   4096
#define DIM     256
#define NROWS   8192                  // 2*BATCH
#define NTILES  512                   // NROWS / 16
#define NSEG    4
#define TILES_PER_SEG (NTILES / NSEG)
#define INV_TEMP 2.0f                 // 1 / 0.5

typedef __attribute__((ext_vector_type(16))) _Float16 v16h;
typedef __attribute__((ext_vector_type(8)))  _Float16 v8h;
typedef __attribute__((ext_vector_type(8)))  float    v8f;
typedef __attribute__((ext_vector_type(4)))  float    v4f;

// ---------------------------------------------------------------------------
// Kernel 1: row-normalize z = concat(z_i, z_j) into f16, one row per wave32.
// ---------------------------------------------------------------------------
__global__ __launch_bounds__(256) void ntx_normalize(const float* __restrict__ zi,
                                                     const float* __restrict__ zj,
                                                     _Float16* __restrict__ zn) {
  const int row  = (blockIdx.x * blockDim.x + threadIdx.x) >> 5;   // 0..8191
  const int lane = threadIdx.x & 31;
  const float* src = (row < BATCH) ? (zi + (size_t)row * DIM)
                                   : (zj + (size_t)(row - BATCH) * DIM);
  v4f x0 = *(const v4f*)(src + lane * 8);
  v4f x1 = *(const v4f*)(src + lane * 8 + 4);
  float ss = x0.x*x0.x + x0.y*x0.y + x0.z*x0.z + x0.w*x0.w
           + x1.x*x1.x + x1.y*x1.y + x1.z*x1.z + x1.w*x1.w;
#pragma unroll
  for (int off = 16; off > 0; off >>= 1) ss += __shfl_xor(ss, off, 32);
  const float inv = 1.0f / sqrtf(ss);
  v8h h;
  h[0] = (_Float16)(x0.x * inv); h[1] = (_Float16)(x0.y * inv);
  h[2] = (_Float16)(x0.z * inv); h[3] = (_Float16)(x0.w * inv);
  h[4] = (_Float16)(x1.x * inv); h[5] = (_Float16)(x1.y * inv);
  h[6] = (_Float16)(x1.z * inv); h[7] = (_Float16)(x1.w * inv);
  *(v8h*)(zn + (size_t)row * DIM + lane * 8) = h;
}

// ---------------------------------------------------------------------------
// B-fragment tile load: 16 x global_load_b128 into a 64-VGPR buffer.
// Column N = l16 -> row (ct*16 + l16) of Zn; half-wave g holds K = [32k+8g..+7] and [32k+16+8g..+7].
// ---------------------------------------------------------------------------
__device__ __forceinline__ void loadB(const _Float16* __restrict__ zn,
                                      int ct, int l16, int g, v16h (&b)[8]) {
  const _Float16* brow = zn + (size_t)(ct * 16 + l16) * DIM;
#pragma unroll
  for (int kc = 0; kc < 8; ++kc) {
    const int kb = kc * 32 + g * 8;
    v8h lo = *(const v8h*)(brow + kb);
    v8h hi = *(const v8h*)(brow + kb + 16);
    b[kc] = __builtin_shufflevector(lo, hi, 0,1,2,3,4,5,6,7,8,9,10,11,12,13,14,15);
  }
}

// ---------------------------------------------------------------------------
// One 16x16 output tile: 8 WMMAs over K=256, fused exp row-sum + pos capture.
// C layout: VGPR v, lane L -> M = v + 8*(L/16), N = L%16.
// ---------------------------------------------------------------------------
__device__ __forceinline__ void tileCompute(const v16h (&a)[8], const v16h (&b)[8],
                                            int ct, int posTile, int lane,
                                            float (&sumExp)[8], float& posVal) {
  v8f c = {};
#pragma unroll
  for (int kc = 0; kc < 8; ++kc)
    c = __builtin_amdgcn_wmma_f32_16x16x32_f16(
        /*neg_a=*/false, a[kc], /*neg_b=*/false, b[kc],
        /*c_mod=*/(short)0, c, /*reuse_a=*/false, /*reuse_b=*/false);
#pragma unroll
  for (int v = 0; v < 8; ++v) sumExp[v] += __expf(INV_TEMP * c[v]);
  if (ct == posTile) {            // wave-uniform branch; capture tile diagonal
#pragma unroll
    for (int v = 0; v < 8; ++v)
      if (lane == v || lane == v + 24) posVal = INV_TEMP * c[v];
  }
}

// ---------------------------------------------------------------------------
// Kernel 2: fused S = Zn*Zn^T / T, denom[r] = sum_j exp(S), positive capture.
// One wave32 owns a 16-row strip; A resident in VGPRs; B double-buffered
// across column tiles so loads of tile n+1 overlap WMMAs of tile n.
// grid = (NTILES/8, NSEG), block = 256 (8 waves).
// ---------------------------------------------------------------------------
__global__ __launch_bounds__(256) void ntx_sim(const _Float16* __restrict__ zn,
                                               float* __restrict__ denomPart,
                                               float* __restrict__ pos) {
  const int lane    = threadIdx.x & 31;
  const int strip   = blockIdx.x * 8 + (threadIdx.x >> 5);  // 0..511
  const int seg     = blockIdx.y;                           // column segment
  const int rowBase = strip * 16;
  const int g       = lane >> 4;                            // half-wave
  const int l16     = lane & 15;

  // A fragments (resident): row = rowBase + l16; half-wave g holds
  // K = [32k + 8g .. +7] and [32k + 16 + 8g .. +7]  (ISA 16-bit A 16x32 layout)
  v16h a[8];
  const _Float16* arow = zn + (size_t)(rowBase + l16) * DIM;
#pragma unroll
  for (int kc = 0; kc < 8; ++kc) {
    const int kb = kc * 32;
    v8h lo = *(const v8h*)(arow + kb + g * 8);
    v8h hi = *(const v8h*)(arow + kb + 16 + g * 8);
    a[kc] = __builtin_shufflevector(lo, hi, 0,1,2,3,4,5,6,7,8,9,10,11,12,13,14,15);
  }

  float sumExp[8];
#pragma unroll
  for (int v = 0; v < 8; ++v) sumExp[v] = 0.0f;
  float posVal = 0.0f;
  const int posTile = (strip < NTILES / 2) ? (strip + NTILES / 2)
                                           : (strip - NTILES / 2);
  const int ct0   = seg * TILES_PER_SEG;
  const int ctEnd = ct0 + TILES_PER_SEG;

  v16h b0[8], b1[8];
  loadB(zn, ct0, l16, g, b0);
  for (int ct = ct0; ct < ctEnd; ct += 2) {
    loadB(zn, ct + 1, l16, g, b1);                  // in flight during tile ct
    tileCompute(a, b0, ct, posTile, lane, sumExp, posVal);
    const int ctn = (ct + 2 < ctEnd) ? (ct + 2) : ct0;   // clamped tail load
    loadB(zn, ctn, l16, g, b0);                     // in flight during tile ct+1
    tileCompute(a, b1, ct + 1, posTile, lane, sumExp, posVal);
  }

  // reduce each row-sum across the 16 lanes (columns) of its half-wave
#pragma unroll
  for (int v = 0; v < 8; ++v) {
    float s = sumExp[v];
    s += __shfl_xor(s, 1, 32);
    s += __shfl_xor(s, 2, 32);
    s += __shfl_xor(s, 4, 32);
    s += __shfl_xor(s, 8, 32);
    sumExp[v] = s;
  }
  if (l16 < 8)   // lane l16 of half-wave g owns row (8g + l16) of the strip
    denomPart[(size_t)seg * NROWS + rowBase + 8 * g + l16] = sumExp[l16];

  if (posTile >= ct0 && posTile < ctEnd) {
    if (lane < 8)        pos[rowBase + lane] = posVal;        // rows 0..7
    else if (lane >= 24) pos[rowBase + lane - 16] = posVal;   // rows 8..15
  }
}

// ---------------------------------------------------------------------------
// Kernel 3: deterministic finalize. denom = sum of 4 segment partials (fixed
// order), loss_r = log(denom - 1) - positive_r, mean over 8192 rows.
// ---------------------------------------------------------------------------
__global__ __launch_bounds__(256) void ntx_finalize(const float* __restrict__ denomPart,
                                                    const float* __restrict__ pos,
                                                    float* __restrict__ out) {
  __shared__ float sdata[256];
  float acc = 0.0f;
  for (int r = threadIdx.x; r < NROWS; r += 256) {
    float d = denomPart[r] + denomPart[NROWS + r] +
              denomPart[2 * NROWS + r] + denomPart[3 * NROWS + r];
    acc += logf(d - 1.0f) - pos[r];
  }
  sdata[threadIdx.x] = acc;
  __syncthreads();
  for (int s = 128; s > 0; s >>= 1) {
    if ((int)threadIdx.x < s) sdata[threadIdx.x] += sdata[threadIdx.x + s];
    __syncthreads();
  }
  if (threadIdx.x == 0) out[0] = sdata[0] * (1.0f / NROWS);
}

// ---------------------------------------------------------------------------
extern "C" void kernel_launch(void* const* d_in, const int* in_sizes, int n_in,
                              void* d_out, int out_size, void* d_ws, size_t ws_size,
                              hipStream_t stream) {
  const float* zi = (const float*)d_in[0];
  const float* zj = (const float*)d_in[1];
  float* out = (float*)d_out;

  char* ws = (char*)d_ws;
  _Float16* zn      = (_Float16*)ws;                               // 4 MB
  float* denomPart  = (float*)(ws + (size_t)NROWS * DIM * 2);      // 128 KB
  float* pos        = (float*)(ws + (size_t)NROWS * DIM * 2
                                  + (size_t)NSEG * NROWS * 4);     // 32 KB

  ntx_normalize<<<NROWS / 8, 256, 0, stream>>>(zi, zj, zn);
  dim3 grid(NTILES / 8, NSEG);
  ntx_sim<<<grid, 256, 0, stream>>>(zn, denomPart, pos);
  ntx_finalize<<<1, 256, 0, stream>>>(denomPart, pos, out);
}
